// AcPredict_71330816852077
// MI455X (gfx1250) — matrix operand order repaired and checked by reading, not compiled, not run on hardware
//
#include <hip/hip_runtime.h>

// ---- shapes from the reference ----
#define LOD   60
#define LSD   120          // 2*LOD
#define AD    10
#define NB    15
#define H     60
#define HP    64           // hidden K padded 60 -> 64 (2 x K32 WMMA steps)
#define OP2   128          // control output cols padded 120 -> 128 (8 N-tiles)
#define NDIAG 7            // band |i-j| <= 3
#define EPM   (LOD*NDIAG)  // 420 band entries per matrix
#define NENT  (4*EPM)      // 1680 real entries (t11,t12,t21,t22)
#define NENTP 1792         // padded to 112 tiles = 8 waves x 14 tiles
#define NTILE (NENTP/16)   // 112
#define TPW   (NTILE/8)    // 14 tiles per wave (compile-time constant)
#define TSTR  1796         // LDS row stride: 8*1796 mod 64 = 32 -> no bank conflict
#define KPAD  32           // coef K padded 15 -> 32
#define BT    16           // batch rows per workgroup (one WMMA M-tile)
#define VROW  300          // staged vector floats per batch row (120+60+60+60)
#define VSEG  (BT*VROW/4)  // 1200 16-byte async segments per workgroup

typedef __attribute__((ext_vector_type(16))) _Float16 v16h;
typedef __attribute__((ext_vector_type(8)))  _Float16 v8h;
typedef __attribute__((ext_vector_type(8)))  float    v8f;

// async global -> LDS 16-byte copy (GLOBAL_LOAD_ASYNC_TO_LDS_B128, ASYNCcnt)
__device__ __forceinline__ void async_copy_b128(const void* lds_dst,
                                                const void* gsrc) {
  unsigned lds_off = (unsigned)(unsigned long long)lds_dst; // low 32b = LDS offset
  asm volatile("global_load_async_to_lds_b128 %0, %1, off"
               :: "v"(lds_off), "v"((unsigned long long)gsrc)
               : "memory");
}

// -------------------------------------------------------------------------
// Kernel 1a: pack the four banded mixing tensors into an f16 B-matrix table.
// Tband[entry][k], entry = mat*420 + i*7 + d (j = i+d-3), k = 0..31.
// k >= 15, out-of-band j, and entries >= 1680 are zero.
// -------------------------------------------------------------------------
__global__ void build_tband_kernel(const float* __restrict__ tm11,
                                   const float* __restrict__ tm12,
                                   const float* __restrict__ tm21,
                                   const float* __restrict__ tm22,
                                   _Float16* __restrict__ tband) {
  int e = blockIdx.x * blockDim.x + threadIdx.x;
  if (e >= NENTP) return;
  _Float16* dst = tband + (size_t)e * KPAD;
  if (e >= NENT) {
    #pragma unroll
    for (int k = 0; k < KPAD; ++k) dst[k] = (_Float16)0.0f;
    return;
  }
  int mat = e / EPM;
  int rem = e % EPM;
  int i   = rem / NDIAG;
  int d   = rem % NDIAG;
  int j   = i + d - 3;
  const float* tm = (mat == 0) ? tm11 : (mat == 1) ? tm12
                  : (mat == 2) ? tm21 : tm22;
  bool inband = (j >= 0) && (j < LOD);
  #pragma unroll
  for (int k = 0; k < KPAD; ++k) {
    float v = (inband && k < NB) ? tm[(size_t)k * LOD * LOD + i * LOD + j] : 0.0f;
    dst[k] = (_Float16)v;
  }
}

// -------------------------------------------------------------------------
// Kernel 1b: pack w_c2 (120x60 f32) -> wc2h[128][64] f16, K-contiguous,
// zero-padded, so the control-GEMM B-fragment is one aligned 32B load.
// -------------------------------------------------------------------------
__global__ void pack_wc2_kernel(const float* __restrict__ w_c2,
                                _Float16* __restrict__ wc2h) {
  int o = blockIdx.x * blockDim.x + threadIdx.x;
  if (o >= OP2) return;
  _Float16* dst = wc2h + (size_t)o * HP;
  #pragma unroll
  for (int k = 0; k < HP; ++k) {
    float v = (o < LSD && k < H) ? w_c2[(size_t)o * H + k] : 0.0f;
    dst[k] = (_Float16)v;
  }
}

// -------------------------------------------------------------------------
// Kernel 2: fused predict for a tile of 16 batch rows, 256 threads (8 waves).
//   P0 async-stage per-batch vectors to LDS (overlaps P1..P5)
//   P1 logits (scalar)        P2 softmax -> coef f16 (A-frag layout in LDS)
//   P3 hidden (scalar, K=10, stored f16)
//   P4 control GEMM via WMMA (K=64, one N-tile per wave)
//   P5 t = coef @ Tband via WMMA (14 tiles/wave, fully unrolled)
//   P6 band consume (LDS only) -> next_mean / ncu / ncl / ncs
// -------------------------------------------------------------------------
__global__ void __launch_bounds__(256)
fused_predict_kernel(const float* __restrict__ post_mean,
                     const float* __restrict__ cu_g,
                     const float* __restrict__ cl_g,
                     const float* __restrict__ cs_g,
                     const float* __restrict__ action,
                     const float* __restrict__ log_noise,
                     const float* __restrict__ w_coef,
                     const float* __restrict__ b_coef,
                     const float* __restrict__ w_c1,
                     const float* __restrict__ b_c1,
                     const float* __restrict__ b_c2,
                     const _Float16* __restrict__ tband,
                     const _Float16* __restrict__ wc2h,
                     float* __restrict__ out,
                     int Bn) {
  extern __shared__ float smem[];
  float*    tL       = smem;                              // [BT][TSTR] f32
  float*    logitsL  = tL + BT * TSTR;                    // [BT][NB]
  float*    controlL = logitsL + BT * NB;                 // [BT][LSD]
  _Float16* coefh    = (_Float16*)(controlL + BT * LSD);  // [BT][KPAD] f16
  _Float16* hiddenh  = coefh + BT * KPAD;                 // [BT][HP]   f16
  float*    vecL     = (float*)(hiddenh + BT * HP);       // [BT][300]  f32

  const int tid   = threadIdx.x;
  const int b0    = blockIdx.x * BT;
  const int lane  = tid & 31;
  const int wv    = tid >> 5;        // 8 wave32s
  const int lhalf = lane >> 4;       // 0: lanes 0-15, 1: lanes 16-31
  const int ln    = lane & 15;

  // ---- P0: async DMA of this tile's vectors into LDS (ASYNCcnt) ----
  // row layout: [0,120) post_mean, [120,180) cu, [180,240) cl, [240,300) cs
  for (int s = tid; s < VSEG; s += 256) {
    int b   = s / (VROW / 4);
    int off = (s % (VROW / 4)) * 4;          // float offset within row
    int bg  = b0 + b;
    if (bg >= Bn) continue;
    const float* src;
    if      (off < 120) src = post_mean + (size_t)bg * LSD + off;
    else if (off < 180) src = cu_g + (size_t)bg * LOD + (off - 120);
    else if (off < 240) src = cl_g + (size_t)bg * LOD + (off - 180);
    else                src = cs_g + (size_t)bg * LOD + (off - 240);
    async_copy_b128(vecL + b * VROW + off, src);
  }

  // warm L2/WGP$ for this wave's first B-fragment while scalar phases run
  __builtin_prefetch(tband + (size_t)(wv * 16 + ln) * KPAD, 0, 1);

  // ---- P1: logits[b][n] = post_mean[b] . w_coef[n] + b_coef[n] ----
  for (int t = tid; t < BT * NB; t += 256) {
    int b = t / NB, n = t % NB;
    float acc = 0.0f;
    if (b0 + b < Bn) {
      const float* pm = post_mean + (size_t)(b0 + b) * LSD;
      const float* w  = w_coef + (size_t)n * LSD;
      acc = b_coef[n];
      #pragma unroll 4
      for (int j = 0; j < LSD; ++j) acc += pm[j] * w[j];
    }
    logitsL[t] = acc;
  }
  __syncthreads();

  // ---- P2: softmax over NB -> coef f16, K padded to 32 ----
  if (tid < BT) {
    bool valid = (b0 + tid) < Bn;
    float ex[NB];
    float inv = 0.0f;
    if (valid) {
      float m = -1e30f;
      for (int n = 0; n < NB; ++n) m = fmaxf(m, logitsL[tid * NB + n]);
      float s = 0.0f;
      for (int n = 0; n < NB; ++n) { ex[n] = __expf(logitsL[tid * NB + n] - m); s += ex[n]; }
      inv = 1.0f / s;
    } else {
      for (int n = 0; n < NB; ++n) ex[n] = 0.0f;
    }
    for (int k = 0; k < KPAD; ++k)
      coefh[tid * KPAD + k] = (_Float16)((k < NB) ? ex[k] * inv : 0.0f);
  }

  // ---- P3: hidden = relu(action @ w_c1^T + b_c1), f16, padded [BT][64] ----
  for (int t = tid; t < BT * HP; t += 256) {
    int b = t / HP, h = t % HP;
    float acc = 0.0f;
    if (h < H && (b0 + b) < Bn) {
      const float* ac = action + (size_t)(b0 + b) * AD;
      acc = b_c1[h];
      #pragma unroll
      for (int a = 0; a < AD; ++a) acc += ac[a] * w_c1[h * AD + a];
      acc = fmaxf(acc, 0.0f);
    }
    hiddenh[t] = (_Float16)acc;
  }
  __syncthreads();   // coefh + hiddenh ready

  // ---- P4: control = hidden @ w_c2^T + b_c2 via WMMA ----
  // A = hidden (16x64 f16 from LDS), B = wc2h (64x128 padded, ws table).
  {
    const int o = wv * 16 + ln;            // output column (o < 120 valid)
    v8f c = {};
    #pragma unroll
    for (int ck = 0; ck < 2; ++ck) {       // K chunks of 32
      // A-frag: element e -> K=ck*32+16*lhalf+e. One 32B LDS load.
      const _Float16* hp = hiddenh + ln * HP + ck * 32 + 16 * lhalf;
      v16h a = *(const v16h*)hp;
      // B-frag: lane N=o, element e -> K = ck*32 + 16*lhalf + e. One 32B load.
      v16h b = *(const v16h*)(wc2h + (size_t)o * HP + ck * 32 + 16 * lhalf);
      c = __builtin_amdgcn_wmma_f32_16x16x32_f16(false, a, false, b,
                                                 (short)0, c, false, false);
    }
    if (o < LSD) {
      float bias = b_c2[o];
      #pragma unroll
      for (int r = 0; r < 8; ++r)          // D: VGPR r -> batch row r+8*lhalf
        controlL[(r + 8 * lhalf) * LSD + o] = c[r] + bias;
    }
  }

  // ---- P5: WMMA  t[b][entry] = sum_k coef[b][k] * Tband[k][entry] ----
  {
    // A-fragment (16x32 f16): lane M=ln, element e -> K=16*lhalf+e
    const _Float16* cp = coefh + ln * KPAD + 16 * lhalf;
    v16h afrag = *(const v16h*)cp;
    // exactly 14 tiles per wave: compile-time loop, wave-uniform, EXEC all-1s
    #pragma unroll
    for (int it = 0; it < TPW; ++it) {
      const int et = wv + 8 * it;
      const int N  = et * 16 + ln;
      v16h bfrag = *(const v16h*)(tband + (size_t)N * KPAD + lhalf * 16);
      v8f c = {};
      c = __builtin_amdgcn_wmma_f32_16x16x32_f16(false, afrag, false, bfrag,
                                                 (short)0, c, false, false);
      const int rowbase = lhalf * 8;
      #pragma unroll
      for (int r = 0; r < 8; ++r)
        tL[(rowbase + r) * TSTR + N] = c[r];
    }
  }
  // all async segments issued by this wave must have landed in LDS
  asm volatile("s_wait_asynccnt 0x0" ::: "memory");
  __syncthreads();

  // ---- P6: band consume (LDS only) -> outputs ----
  const size_t covBase = (size_t)Bn * LSD;
  for (int t = tid; t < BT * LOD; t += 256) {
    int b = t / LOD, i = t % LOD;
    int bg = b0 + b;
    if (bg >= Bn) continue;
    const float* pm   = vecL + b * VROW;        // [0,120)
    const float* cuv  = pm + 120;               // [120,180)
    const float* clv  = pm + 180;               // [180,240)
    const float* csv  = pm + 240;               // [240,300)
    const float* trow = tL + b * TSTR;

    float nmu = 0.f, nml = 0.f, ncu = 0.f, ncl = 0.f, ncs = 0.f;
    #pragma unroll
    for (int d = 0; d < NDIAG; ++d) {
      int j = i + d - 3;
      if (j < 0 || j >= LOD) continue;
      float eye = (d == 3) ? 1.0f : 0.0f;
      float t11 = trow[0 * EPM + i * NDIAG + d] + eye;
      float t12 = trow[1 * EPM + i * NDIAG + d];
      float t21 = trow[2 * EPM + i * NDIAG + d];
      float t22 = trow[3 * EPM + i * NDIAG + d] + eye;
      float muj = pm[j], mlj = pm[LOD + j];
      float cuj = cuv[j], clj = clv[j], csj = csv[j];
      nmu += t11 * muj + t12 * mlj;
      nml += t21 * muj + t22 * mlj;
      ncu += t11 * t11 * cuj + 2.0f * t11 * t12 * csj + t12 * t12 * clj;
      ncl += t21 * t21 * cuj + 2.0f * t21 * t22 * csj + t22 * t22 * clj;
      ncs += t21 * t11 * cuj + (t22 * t11 + t21 * t12) * csj + t22 * t12 * clj;
    }
    float lnu = log_noise[i], lnl = log_noise[LOD + i];
    ncu += (lnu < 0.0f) ? __expf(lnu) : (lnu + 1.0f);   // elup1
    ncl += (lnl < 0.0f) ? __expf(lnl) : (lnl + 1.0f);

    out[(size_t)bg * LSD + i]       = nmu + controlL[b * LSD + i];
    out[(size_t)bg * LSD + LOD + i] = nml + controlL[b * LSD + LOD + i];
    out[covBase + (size_t)bg * LOD + i]                        = ncu;
    out[covBase + (size_t)Bn * LOD + (size_t)bg * LOD + i]     = ncl;
    out[covBase + (size_t)2 * Bn * LOD + (size_t)bg * LOD + i] = ncs;
  }
}

// -------------------------------------------------------------------------
extern "C" void kernel_launch(void* const* d_in, const int* in_sizes, int n_in,
                              void* d_out, int out_size, void* d_ws, size_t ws_size,
                              hipStream_t stream) {
  // setup_inputs() order:
  const float* post_mean = (const float*)d_in[0];
  const float* cu        = (const float*)d_in[1];
  const float* cl        = (const float*)d_in[2];
  const float* cs        = (const float*)d_in[3];
  const float* action    = (const float*)d_in[4];
  const float* tm11      = (const float*)d_in[5];
  const float* tm12      = (const float*)d_in[6];
  const float* tm21      = (const float*)d_in[7];
  const float* tm22      = (const float*)d_in[8];
  const float* log_noise = (const float*)d_in[9];
  const float* w_coef    = (const float*)d_in[10];
  const float* b_coef    = (const float*)d_in[11];
  const float* w_c1      = (const float*)d_in[12];
  const float* b_c1      = (const float*)d_in[13];
  const float* w_c2      = (const float*)d_in[14];
  const float* b_c2      = (const float*)d_in[15];

  const int Bn = in_sizes[0] / LSD;   // 8192

  _Float16* tband = (_Float16*)d_ws;               // NENTP*32 f16 = 112 KB
  _Float16* wc2h  = tband + (size_t)NENTP * KPAD;  // 128*64 f16  =  16 KB

  build_tband_kernel<<<(NENTP + 255) / 256, 256, 0, stream>>>(
      tm11, tm12, tm21, tm22, tband);
  pack_wc2_kernel<<<1, OP2, 0, stream>>>(w_c2, wc2h);

  const int smemBytes =
      (BT * TSTR + BT * NB + BT * LSD + BT * VROW) * (int)sizeof(float)
      + (BT * KPAD + BT * HP) * (int)sizeof(_Float16);   // ~145.9 KB dyn LDS
  fused_predict_kernel<<<(Bn + BT - 1) / BT, 256, smemBytes, stream>>>(
      post_mean, cu, cl, cs, action, log_noise,
      w_coef, b_coef, w_c1, b_c1, b_c2,
      tband, wc2h, (float*)d_out, Bn);
}
